// NeuralSATSolver_37864431681664
// MI455X (gfx1250) — compile-verified, matrix operand order checked by reading, not run on hardware
//
#include <hip/hip_runtime.h>
#include <hip/hip_bf16.h>
#include <math.h>

#define H 128
#define NP 50000
#define NC 20000
#define KC 3
#define NB 4

typedef __attribute__((ext_vector_type(16))) __bf16 v16bf;
typedef __attribute__((ext_vector_type(8)))  __bf16 v8bf;
typedef __attribute__((ext_vector_type(8)))  float  v8f;

// ---------------------------------------------------------------------------
// Load a 16x32 bf16 WMMA fragment (A or B operand) from a row-major bf16 tile.
// CDNA5 16-bit A layout (ISA 7.12.2):
//   lanes 0-15 : row = lane,    elems 0-7 = K0..7,   elems 8-15 = K16..23
//   lanes16-31 : row = lane-16, elems 0-7 = K8..15,  elems 8-15 = K24..31
// B (32x16) mirrors this with lane = column index.  Two 16B loads per lane.
// ---------------------------------------------------------------------------
__device__ inline v16bf load_frag_bf16(const __bf16* __restrict__ tile, int ld) {
  const int lane = threadIdx.x & 31;
  const int r  = lane & 15;
  const int ko = (lane >> 4) << 3;           // 0 or 8
  const __bf16* base = tile + (size_t)r * ld + ko;
  v8bf c0 = *(const v8bf*)(base);
  v8bf c1 = *(const v8bf*)(base + 16);
  v16bf f;
#pragma unroll
  for (int j = 0; j < 8; ++j) { f[j] = c0[j]; f[8 + j] = c1[j]; }
  return f;
}

// Y[rows,128] = X[rows,128] @ W[128,128]^T + bias   (out[n] = sum_k X[k]*W[n,k])
// 8 waves / block, each wave computes a 16x128 stripe with 32 v_wmma ops.
__global__ __launch_bounds__(256)
void gemm128_bias_kernel(const __bf16* __restrict__ X, const __bf16* __restrict__ W,
                         const float* __restrict__ bias, float* __restrict__ Y,
                         int rows) {
  const int wave = threadIdx.x >> 5;
  const int row0 = blockIdx.x * 128 + wave * 16;
  if (row0 >= rows) return;                  // wave-uniform: EXEC stays all-1s
  const int lane = threadIdx.x & 31;
  const int col  = lane & 15;
  const int mhi  = (lane >> 4) * 8;

  v16bf a[4];
#pragma unroll
  for (int kt = 0; kt < 4; ++kt)
    a[kt] = load_frag_bf16(X + (size_t)row0 * H + kt * 32, H);

#pragma unroll
  for (int nt = 0; nt < 8; ++nt) {
    v8f acc = {};
#pragma unroll
    for (int kt = 0; kt < 4; ++kt) {
      v16bf b = load_frag_bf16(W + (size_t)(nt * 16) * H + kt * 32, H);
      acc = __builtin_amdgcn_wmma_f32_16x16x32_bf16(false, a[kt], false, b,
                                                    (short)0, acc, false, false);
    }
    const float bn = bias[nt * 16 + col];
#pragma unroll
    for (int i = 0; i < 8; ++i)
      Y[(size_t)(row0 + mhi + i) * H + nt * 16 + col] = acc[i] + bn;
  }
}

// Fused readout: out[row] = sigmoid( sum_n relu(X@W1^T + b1)[row,n]*w2[n] + b2 )
__global__ __launch_bounds__(256)
void readout_kernel(const __bf16* __restrict__ X, const __bf16* __restrict__ W1,
                    const float* __restrict__ b1, const float* __restrict__ w2,
                    const float* __restrict__ b2, float* __restrict__ out,
                    int rows) {
  const int wave = threadIdx.x >> 5;
  const int row0 = blockIdx.x * 128 + wave * 16;
  if (row0 >= rows) return;
  const int lane = threadIdx.x & 31;
  const int col  = lane & 15;
  const int mhi  = (lane >> 4) * 8;

  v16bf a[4];
#pragma unroll
  for (int kt = 0; kt < 4; ++kt)
    a[kt] = load_frag_bf16(X + (size_t)row0 * H + kt * 32, H);

  float part[8];
#pragma unroll
  for (int i = 0; i < 8; ++i) part[i] = 0.f;

#pragma unroll
  for (int nt = 0; nt < 8; ++nt) {
    v8f acc = {};
#pragma unroll
    for (int kt = 0; kt < 4; ++kt) {
      v16bf b = load_frag_bf16(W1 + (size_t)(nt * 16) * H + kt * 32, H);
      acc = __builtin_amdgcn_wmma_f32_16x16x32_bf16(false, a[kt], false, b,
                                                    (short)0, acc, false, false);
    }
    const int   n  = nt * 16 + col;
    const float bn = b1[n];
    const float wn = w2[n];
#pragma unroll
    for (int i = 0; i < 8; ++i) {
      float t = acc[i] + bn;
      t = t > 0.f ? t : 0.f;
      part[i] += t * wn;
    }
  }
  // reduce the 16 columns held by each 16-lane group
#pragma unroll
  for (int m = 8; m >= 1; m >>= 1)
#pragma unroll
    for (int i = 0; i < 8; ++i) part[i] += __shfl_xor(part[i], m, 16);

  if (col == 0) {
    const float bb = b2[0];
#pragma unroll
    for (int i = 0; i < 8; ++i) {
      float x = part[i] + bb;
      out[row0 + mhi + i] = 1.f / (1.f + expf(-x));
    }
  }
}

// g[bc,:] = mean_k vs[b, idx[bc,k], :]   (written directly as bf16)
__global__ void gather_mean_kernel(const int* __restrict__ idx,
                                   const float* __restrict__ vs,
                                   __bf16* __restrict__ g) {
  const int bc = blockIdx.x;
  const int h  = threadIdx.x;
  const int b  = bc / NC;
  const int i0 = idx[bc * KC + 0];
  const int i1 = idx[bc * KC + 1];
  const int i2 = idx[bc * KC + 2];
  const float* vb = vs + (size_t)b * NP * H;
  float s = vb[(size_t)i0 * H + h] + vb[(size_t)i1 * H + h] + vb[(size_t)i2 * H + h];
  g[(size_t)bc * H + h] = (__bf16)(s * (1.f / 3.f));
}

// counts[p] += 1 for each first occurrence inside a clause (summed over batch).
// One thread per clause: K=3, so in-clause dedup is two index compares.
__global__ void counts_kernel(const int* __restrict__ idx, float* __restrict__ counts,
                              int nclauses) {
  int bc = blockIdx.x * blockDim.x + threadIdx.x;
  if (bc >= nclauses) return;
  const int i0 = idx[bc * KC + 0];
  const int i1 = idx[bc * KC + 1];
  const int i2 = idx[bc * KC + 2];
  atomicAdd(&counts[i0], 1.0f);
  if (i1 != i0)             atomicAdd(&counts[i1], 1.0f);
  if (i2 != i0 && i2 != i1) atomicAdd(&counts[i2], 1.0f);
}

// sums[v,:] += M[bc,:] for each first occurrence.  One block per clause:
// the 512B message row is loaded once and fanned out to <=3 predicates.
__global__ void scatter_kernel(const int* __restrict__ idx,
                               const float* __restrict__ M,
                               float* __restrict__ sums) {
  const int bc = blockIdx.x;
  const int h  = threadIdx.x;
  const int i0 = idx[bc * KC + 0];
  const int i1 = idx[bc * KC + 1];
  const int i2 = idx[bc * KC + 2];
  const float m = M[(size_t)bc * H + h];
  atomicAdd(&sums[(size_t)i0 * H + h], m);
  if (i1 != i0)             atomicAdd(&sums[(size_t)i1 * H + h], m);
  if (i2 != i0 && i2 != i1) atomicAdd(&sums[(size_t)i2 * H + h], m);
}

// One thread per (p,h): compute the broadcast message once, apply to all B
// batch slices (message is batch-invariant in the reference).
__global__ void update_f32_kernel(const float* __restrict__ vin,
                                  const float* __restrict__ sums,
                                  const float* __restrict__ counts,
                                  float* __restrict__ vout) {
  size_t t = (size_t)blockIdx.x * blockDim.x + threadIdx.x;
  const size_t PH = (size_t)NP * H;
  if (t >= PH) return;
  size_t p = t / H;
  float  c = counts[p];
  float  msg = c > 0.f ? sums[t] / fmaxf(c, 1.f) : 0.f;
#pragma unroll
  for (int b = 0; b < NB; ++b)
    vout[(size_t)b * PH + t] = vin[(size_t)b * PH + t] + msg;
}

// Same, but the result is only ever consumed by the bf16 readout GEMM.
__global__ void update_bf16_kernel(const float* __restrict__ vin,
                                   const float* __restrict__ sums,
                                   const float* __restrict__ counts,
                                   __bf16* __restrict__ vout) {
  size_t t = (size_t)blockIdx.x * blockDim.x + threadIdx.x;
  const size_t PH = (size_t)NP * H;
  if (t >= PH) return;
  size_t p = t / H;
  float  c = counts[p];
  float  msg = c > 0.f ? sums[t] / fmaxf(c, 1.f) : 0.f;
#pragma unroll
  for (int b = 0; b < NB; ++b)
    vout[(size_t)b * PH + t] = (__bf16)(vin[(size_t)b * PH + t] + msg);
}

__global__ void zero_kernel(float* __restrict__ p, size_t n) {
  size_t t = (size_t)blockIdx.x * blockDim.x + threadIdx.x;
  if (t < n) p[t] = 0.f;
}

__global__ void f32_to_bf16_kernel(const float* __restrict__ in,
                                   __bf16* __restrict__ out, size_t n) {
  size_t t = (size_t)blockIdx.x * blockDim.x + threadIdx.x;
  if (t < n) out[t] = (__bf16)in[t];
}

// Cm[i,j] = sum_k A[i,k]*Bm[k,j]   (128x128x128, tiny precompute)
__global__ void matmul128_kernel(const float* __restrict__ A,
                                 const float* __restrict__ Bm,
                                 float* __restrict__ Cm) {
  int i = blockIdx.x, j = threadIdx.x;
  float s = 0.f;
  for (int k = 0; k < H; ++k) s += A[i * H + k] * Bm[k * H + j];
  Cm[i * H + j] = s;
}

// y[i] = sum_k W[i,k]*x[k] + b[i]
__global__ void matvec128_kernel(const float* __restrict__ W,
                                 const float* __restrict__ x,
                                 const float* __restrict__ b,
                                 float* __restrict__ y) {
  int i = threadIdx.x;
  float s = b[i];
  for (int k = 0; k < H; ++k) s += W[i * H + k] * x[k];
  y[i] = s;
}

extern "C" void kernel_launch(void* const* d_in, const int* in_sizes, int n_in,
                              void* d_out, int out_size, void* d_ws, size_t ws_size,
                              hipStream_t stream) {
  const int*   idx  = (const int*)d_in[0];
  const float* vs0  = (const float*)d_in[1];
  const float* W_vc = (const float*)d_in[2];
  const float* b_vc = (const float*)d_in[3];
  const float* W_ce = (const float*)d_in[4];
  const float* b_ce = (const float*)d_in[5];
  const float* W_cv = (const float*)d_in[6];
  const float* b_cv = (const float*)d_in[7];
  const float* W_s1 = (const float*)d_in[8];
  const float* b_s1 = (const float*)d_in[9];
  const float* W_s2 = (const float*)d_in[10];
  const float* b_s2 = (const float*)d_in[11];
  float* out = (float*)d_out;

  char* wsb = (char*)d_ws;
  size_t off = 0;
  auto alloc = [&](size_t bytes) -> void* {
    void* p = (void*)(wsb + off);
    off = (off + bytes + 255) & ~(size_t)255;
    return p;
  };
  const size_t BPH = (size_t)NB * NP * H;
  const size_t BCH = (size_t)NB * NC * H;
  const size_t PH  = (size_t)NP * H;
  const size_t HH  = (size_t)H * H;

  float*  vs    = (float*)alloc(BPH * 4);   // evolving f32 state (iter0 -> iter1)
  __bf16* vsbf  = (__bf16*)alloc(BPH * 2);  // final state, bf16, feeds readout
  __bf16* g     = (__bf16*)alloc(BCH * 2);  // per-clause mean (bf16 GEMM input)
  float*  M     = (float*)alloc(BCH * 4);   // clause->variable messages
  float*  sums  = (float*)alloc(PH * 4);    // segment sums [P,H]
  float*  cnts  = (float*)alloc((size_t)NP * 4);
  float*  T1    = (float*)alloc(HH * 4);
  float*  Wcomb = (float*)alloc(HH * 4);
  __bf16* Wcbf  = (__bf16*)alloc(HH * 2);   // bf16 combined clause weight
  __bf16* Ws1bf = (__bf16*)alloc(HH * 2);   // bf16 readout weight
  float*  btmp  = (float*)alloc(512);
  float*  bcomb = (float*)alloc(512);

  const int BC  = NB * NC;                  // 80000
  const int BP  = NB * NP;                  // 200000

  // --- counts (once) ---
  zero_kernel<<<(NP + 255) / 256, 256, 0, stream>>>(cnts, (size_t)NP);
  counts_kernel<<<(BC + 255) / 256, 256, 0, stream>>>(idx, cnts, BC);

  // --- fold Wcv@Wce@Wvc into one 128x128 matrix + bias; convert weights once ---
  matmul128_kernel<<<H, H, 0, stream>>>(W_ce, W_vc, T1);     // T1 = Wce @ Wvc
  matmul128_kernel<<<H, H, 0, stream>>>(W_cv, T1, Wcomb);    // Wcomb = Wcv @ T1
  matvec128_kernel<<<1, H, 0, stream>>>(W_ce, b_vc, b_ce, btmp);
  matvec128_kernel<<<1, H, 0, stream>>>(W_cv, btmp, b_cv, bcomb);
  f32_to_bf16_kernel<<<(int)((HH + 255) / 256), 256, 0, stream>>>(Wcomb, Wcbf, HH);
  f32_to_bf16_kernel<<<(int)((HH + 255) / 256), 256, 0, stream>>>(W_s1, Ws1bf, HH);

  // --- iteration 0: read input state, write f32 state ---
  gather_mean_kernel<<<BC, H, 0, stream>>>(idx, vs0, g);
  gemm128_bias_kernel<<<(BC + 127) / 128, 256, 0, stream>>>(g, Wcbf, bcomb, M, BC);
  zero_kernel<<<(int)((PH + 255) / 256), 256, 0, stream>>>(sums, PH);
  scatter_kernel<<<BC, H, 0, stream>>>(idx, M, sums);
  update_f32_kernel<<<(int)((PH + 255) / 256), 256, 0, stream>>>(vs0, sums, cnts, vs);

  // --- iteration 1: read f32 state, write bf16 state (consumed only by readout) ---
  gather_mean_kernel<<<BC, H, 0, stream>>>(idx, vs, g);
  gemm128_bias_kernel<<<(BC + 127) / 128, 256, 0, stream>>>(g, Wcbf, bcomb, M, BC);
  zero_kernel<<<(int)((PH + 255) / 256), 256, 0, stream>>>(sums, PH);
  scatter_kernel<<<BC, H, 0, stream>>>(idx, M, sums);
  update_bf16_kernel<<<(int)((PH + 255) / 256), 256, 0, stream>>>(vs, sums, cnts, vsbf);

  // --- readout ---
  readout_kernel<<<(BP + 127) / 128, 256, 0, stream>>>(vsbf, Ws1bf, b_s1, W_s2, b_s2,
                                                       out, BP);
}
